// InternLM3CrossAttention_68186900791298
// MI455X (gfx1250) — matrix operand order repaired and checked by reading, not compiled, not run on hardware
//
#include <hip/hip_runtime.h>

typedef __bf16 bf16_t;
typedef __attribute__((ext_vector_type(16))) __bf16 v16bf;
typedef __attribute__((ext_vector_type(8)))  float  v8f;

#define DEVFN __device__ __forceinline__

constexpr int B_ = 8, SKV = 8192, HIDDEN = 4096, NH = 32, NKV = 8, HD = 128, G = 4;
constexpr int CHUNK = 128, NCHUNK = SKV / CHUNK;              // 64 chunks -> 4096 waves
constexpr float SCALE = 0.08838834764831845f;                 // 1/sqrt(128)

// Native f32 -> bf16 conversion: lowers to v_cvt_pk_bf16_f32 on gfx1250.
DEVFN bf16_t f2bf(float f) { return (bf16_t)f; }
DEVFN bf16_t bf_zero() { return __builtin_bit_cast(bf16_t, (unsigned short)0); }

DEVFN v8f wmma_bf16(v16bf a, v16bf b, v8f c) {
  // (neg_a, A, neg_b, B, c_mod, C, reuse_a, reuse_b)
  return __builtin_amdgcn_wmma_f32_16x16x32_bf16(false, a, false, b, (short)0, c,
                                                 false, false);
}

// ---------------------------------------------------------------------------
// C[8 x N] = A[8 x K] * Bm[K x N]   (one wave per 16-column tile)
// A-fragment rows = batch (8 real rows, 8 zero rows); bf16 WMMA, f32 accum.
// ---------------------------------------------------------------------------
__global__ void gemm8_wmma(const float* __restrict__ A,
                           const float* __restrict__ Bm,
                           float* __restrict__ C,
                           int Kdim, int Ncols) {
  const int lane = threadIdx.x & 31;
  const int half = lane >> 4;          // 0: lanes 0-15, 1: lanes 16-31
  const int r    = lane & 15;
  const int col0 = blockIdx.x * 16;

  v8f acc = {};
  for (int k0 = 0; k0 < Kdim; k0 += 32) {
    // A fragment (16-bit A 16x32 layout): elem i -> K = (i>>3)*16 + half*8 + (i&7)
    v16bf af;
    if (r < 8) {
      const float* ap = A + (size_t)r * Kdim + k0 + half * 8;
#pragma unroll
      for (int i = 0; i < 8; ++i) {
        af[i]     = f2bf(ap[i]);
        af[i + 8] = f2bf(ap[i + 16]);
      }
    } else {
#pragma unroll
      for (int i = 0; i < 16; ++i) af[i] = bf_zero();
    }
    // B fragment (16-bit B 32x16): lane holds col N=r, K = k0 + (i>>3)*16 + half*8 + (i&7)
    const float* bp = Bm + (size_t)(k0 + half * 8) * Ncols + (col0 + r);
    __builtin_prefetch(bp + (size_t)32 * Ncols, 0, 0);
    v16bf bf;
#pragma unroll
    for (int i = 0; i < 16; ++i) bf[i] = f2bf(bp[(size_t)((i & 7) + (i >> 3) * 16) * Ncols]);

    acc = wmma_bf16(af, bf, acc);
  }
  // C/D layout: VGPR v, lane -> row = v + half*8, col = col0 + r
  if (half == 0) {
#pragma unroll
    for (int v = 0; v < 8; ++v) C[(size_t)v * Ncols + col0 + r] = acc[v];
  }
}

// ---------------------------------------------------------------------------
// RoPE for q (raw q-proj in ws) and the new K token.
// ---------------------------------------------------------------------------
__global__ void rope_kernel(const float* __restrict__ q_raw,
                            const float* __restrict__ key_states,  // [B][1][NKV][HD]
                            const float* __restrict__ cosb,        // [B][1][HD]
                            const float* __restrict__ sinb,
                            float* __restrict__ q_rope,
                            float* __restrict__ k_rope) {
  int idx = blockIdx.x * blockDim.x + threadIdx.x;
  const int nq = B_ * NH * HD;
  if (idx < nq) {
    int d = idx & (HD - 1);
    int b = idx / (NH * HD);
    float c = cosb[b * HD + d], s = sinb[b * HD + d];
    float x = q_raw[idx];
    float o = (d < 64) ? -q_raw[idx + 64] : q_raw[idx - 64];
    q_rope[idx] = x * c + o * s;
  } else {
    int j = idx - nq;
    if (j < B_ * NKV * HD) {
      int d = j & (HD - 1);
      int b = j / (NKV * HD);
      float c = cosb[b * HD + d], s = sinb[b * HD + d];
      float x = key_states[j];
      float o = (d < 64) ? -key_states[j + 64] : key_states[j - 64];
      k_rope[j] = x * c + o * s;
    }
  }
}

// ---------------------------------------------------------------------------
// Flash-decode partials: one wave per (b, kv-head, 128-position chunk).
// scores = Q(4x128) x K_chunk^T  -> softmax stats -> P x V_chunk partial out.
// ---------------------------------------------------------------------------
__global__ void attn_partial(const float* __restrict__ q_rope,   // [B][NH][HD]
                             const float* __restrict__ k_cache,  // [B][SKV][NKV][HD]
                             const float* __restrict__ v_cache,
                             float* __restrict__ part_o,         // [B][NKV][NCHUNK][G][HD]
                             float* __restrict__ part_m,         // [B][NKV][NCHUNK][G]
                             float* __restrict__ part_l) {
  __shared__ float  s_s[G][CHUNK];
  __shared__ bf16_t s_p[G][CHUNK];

  const int lane = threadIdx.x & 31;
  const int half = lane >> 4;
  const int r    = lane & 15;
  const int chunk = blockIdx.x, n = blockIdx.y, b = blockIdx.z;
  const int pos0 = chunk * CHUNK;
  const size_t krow = (size_t)NKV * HD;   // stride (floats) between positions

  // --- Q fragments: rows = g (4 real rows), K = head-dim, 4 k-steps of 32 ---
  v16bf aq[4];
#pragma unroll
  for (int ks = 0; ks < 4; ++ks) {
    if (r < G) {
      const float* qp = q_rope + ((size_t)(b * NH + n * G + r)) * HD + ks * 32 + half * 8;
#pragma unroll
      for (int i = 0; i < 8; ++i) {
        aq[ks][i]     = f2bf(qp[i]);
        aq[ks][i + 8] = f2bf(qp[i + 16]);
      }
    } else {
#pragma unroll
      for (int i = 0; i < 16; ++i) aq[ks][i] = bf_zero();
    }
  }

  // --- scores: CHUNK/16 tiles of 16 positions, 4 accumulated WMMAs over d=128 ---
  for (int t = 0; t < CHUNK / 16; ++t) {
    const int pos = pos0 + t * 16 + r;
    const float* kp = k_cache + ((size_t)b * SKV + pos) * krow + n * HD;
    __builtin_prefetch(kp + 16 * krow, 0, 0);
    v8f acc = {};
#pragma unroll
    for (int ks = 0; ks < 4; ++ks) {
      const float* kpp = kp + ks * 32 + half * 8;   // two runs of 8 floats, 16 apart
      v16bf bk;
#pragma unroll
      for (int i = 0; i < 16; ++i) bk[i] = f2bf(kpp[(i & 7) + (i >> 3) * 16]);
      acc = wmma_bf16(aq[ks], bk, acc);
    }
    if (half == 0) {
#pragma unroll
      for (int v = 0; v < G; ++v) s_s[v][t * 16 + r] = acc[v] * SCALE;
    }
  }
  __syncthreads();

  // --- chunk-local softmax stats (each lane reduces a stride-32 slice) ---
  float m[G], l[G];
#pragma unroll
  for (int g = 0; g < G; ++g) {
    float mx = -1e30f;
#pragma unroll
    for (int j = 0; j < CHUNK / 32; ++j) mx = fmaxf(mx, s_s[g][lane + 32 * j]);
#pragma unroll
    for (int off = 16; off > 0; off >>= 1) mx = fmaxf(mx, __shfl_xor(mx, off, 32));
    float sum = 0.f;
#pragma unroll
    for (int j = 0; j < CHUNK / 32; ++j) {
      float p = __expf(s_s[g][lane + 32 * j] - mx);
      s_p[g][lane + 32 * j] = f2bf(p);
      sum += p;
    }
#pragma unroll
    for (int off = 16; off > 0; off >>= 1) sum += __shfl_xor(sum, off, 32);
    m[g] = mx;
    l[g] = sum;
  }
  __syncthreads();

  // --- P x V: K-dim = positions (CHUNK/32 steps), N-dim = d (8 tiles of 16) ---
  v8f acc_o[8];
#pragma unroll
  for (int dt = 0; dt < 8; ++dt) acc_o[dt] = (v8f){};

  for (int kstep = 0; kstep < CHUNK / 32; ++kstep) {
    v16bf ap;  // A = P, row = g, K = position offset
#pragma unroll
    for (int i = 0; i < 16; ++i) {
      int kk = kstep * 32 + (i >> 3) * 16 + half * 8 + (i & 7);
      ap[i] = (r < G) ? s_p[r][kk] : bf_zero();
    }
    const int posb = pos0 + kstep * 32 + half * 8;
    const float* vbase = v_cache + ((size_t)b * SKV + posb) * krow + n * HD;
    __builtin_prefetch(vbase + 32 * krow, 0, 0);
#pragma unroll
    for (int dt = 0; dt < 8; ++dt) {
      const float* vp = vbase + dt * 16 + r;   // lane: col d, 16 strided positions
      v16bf bv;
#pragma unroll
      for (int i = 0; i < 16; ++i) bv[i] = f2bf(vp[(size_t)((i & 7) + (i >> 3) * 16) * krow]);
      acc_o[dt] = wmma_bf16(ap, bv, acc_o[dt]);
    }
  }

  // --- write partials ---
  float* po = part_o + (((size_t)(b * NKV + n) * NCHUNK + chunk) * G) * HD;
  if (half == 0) {
#pragma unroll
    for (int dt = 0; dt < 8; ++dt)
#pragma unroll
      for (int v = 0; v < G; ++v) po[(size_t)v * HD + dt * 16 + r] = acc_o[dt][v];
  }
  if (lane == 0) {
    size_t mi = ((size_t)(b * NKV + n) * NCHUNK + chunk) * G;
#pragma unroll
    for (int g = 0; g < G; ++g) { part_m[mi + g] = m[g]; part_l[mi + g] = l[g]; }
  }
}

// ---------------------------------------------------------------------------
// Combine chunk partials + the single appended new token (logsumexp merge).
// One wave per (b, n, g); lanes cover d.
// ---------------------------------------------------------------------------
__global__ void attn_combine(const float* __restrict__ q_rope,
                             const float* __restrict__ k_new,        // roped [B][NKV][HD]
                             const float* __restrict__ value_states, // [B][1][NKV][HD]
                             const float* __restrict__ part_o,
                             const float* __restrict__ part_m,
                             const float* __restrict__ part_l,
                             float* __restrict__ attn_out) {          // [B][NH][HD]
  const int b = blockIdx.x >> 3, n = blockIdx.x & 7;
  const int g = threadIdx.x >> 5, lane = threadIdx.x & 31;

  const float* q  = q_rope + ((size_t)(b * NH + n * G + g)) * HD;
  const float* kn = k_new + ((size_t)(b * NKV + n)) * HD;
  float part = 0.f;
#pragma unroll
  for (int j = 0; j < 4; ++j) part += q[lane + 32 * j] * kn[lane + 32 * j];
#pragma unroll
  for (int off = 16; off > 0; off >>= 1) part += __shfl_xor(part, off, 32);
  const float s_new = part * SCALE;

  const size_t base = ((size_t)(b * NKV + n) * NCHUNK) * G + g;
  float M = s_new;
  for (int c = 0; c < NCHUNK; ++c) M = fmaxf(M, part_m[base + (size_t)c * G]);
  const float wnew = __expf(s_new - M);
  float L = wnew;
  for (int c = 0; c < NCHUNK; ++c)
    L += part_l[base + (size_t)c * G] * __expf(part_m[base + (size_t)c * G] - M);
  const float inv = 1.f / L;

  const float* vn = value_states + ((size_t)(b * NKV + n)) * HD;
#pragma unroll
  for (int j = 0; j < 4; ++j) {
    const int d = lane + 32 * j;
    float acc = wnew * vn[d];
    for (int c = 0; c < NCHUNK; ++c) {
      float w = __expf(part_m[base + (size_t)c * G] - M);
      acc += part_o[(((size_t)(b * NKV + n) * NCHUNK + c) * G + g) * HD + d] * w;
    }
    attn_out[((size_t)(b * NH + n * G + g)) * HD + d] = acc * inv;
  }
}

// ---------------------------------------------------------------------------
extern "C" void kernel_launch(void* const* d_in, const int* in_sizes, int n_in,
                              void* d_out, int out_size, void* d_ws, size_t ws_size,
                              hipStream_t stream) {
  const float* hs    = (const float*)d_in[0];
  const float* wq    = (const float*)d_in[1];
  const float* wo    = (const float*)d_in[2];
  const float* cosb  = (const float*)d_in[3];
  const float* sinb  = (const float*)d_in[4];
  const float* knew  = (const float*)d_in[5];
  const float* vnew  = (const float*)d_in[6];
  const float* kc    = (const float*)d_in[7];
  const float* vc    = (const float*)d_in[8];

  float* ws = (float*)d_ws;
  float* q_raw  = ws;                               // 32768
  float* q_rope = q_raw + 32768;                    // 32768
  float* k_rope = q_rope + 32768;                   // 8192
  float* part_o = k_rope + 8192;                    // B*NKV*NCHUNK*G*HD = 2,097,152
  float* part_m = part_o + (size_t)B_ * NKV * NCHUNK * G * HD;  // 16384
  float* part_l = part_m + B_ * NKV * NCHUNK * G;               // 16384
  float* attn   = part_l + B_ * NKV * NCHUNK * G;               // 32768

  // 1) q = hidden @ wq  (WMMA)
  gemm8_wmma<<<dim3(HIDDEN / 16), 32, 0, stream>>>(hs, wq, q_raw, HIDDEN, NH * HD);
  // 2) RoPE on q and the new K token
  rope_kernel<<<dim3((B_ * NH * HD + B_ * NKV * HD + 255) / 256), 256, 0, stream>>>(
      q_raw, knew, cosb, sinb, q_rope, k_rope);
  // 3) flash-decode partials over the cached 8192 positions (WMMA)
  attn_partial<<<dim3(NCHUNK, NKV, B_), 32, 0, stream>>>(q_rope, kc, vc,
                                                         part_o, part_m, part_l);
  // 4) merge partials + appended token
  attn_combine<<<dim3(B_ * NKV), G * 32, 0, stream>>>(q_rope, k_rope, vnew,
                                                      part_o, part_m, part_l, attn);
  // 5) out = attn @ wo  (WMMA)
  gemm8_wmma<<<dim3(HIDDEN / 16), 32, 0, stream>>>(attn, wo, (float*)d_out,
                                                   NH * HD, HIDDEN);
}
